// Network_35158602285224
// MI455X (gfx1250) — matrix operand order repaired and checked by reading, not compiled, not run on hardware
//
#include <hip/hip_runtime.h>
#include <hip/hip_bf16.h>
#include <cstdint>
#include <cstddef>

typedef __attribute__((ext_vector_type(16))) _Float16 v16h;
typedef __attribute__((ext_vector_type(8)))  _Float16 v8h;
typedef __attribute__((ext_vector_type(8)))  float    v8f;

// ---------------------------------------------------------------------------
// codes[i] = cm + en * exp(cv)   (4 codebooks packed: 160 | 1000 | 1000 | 160)
// ---------------------------------------------------------------------------
__global__ void codes_kernel(const float* __restrict__ cm0, const float* __restrict__ cv0, const float* __restrict__ en0,
                             const float* __restrict__ cm1, const float* __restrict__ cv1, const float* __restrict__ en1,
                             const float* __restrict__ cm2, const float* __restrict__ cv2, const float* __restrict__ en2,
                             const float* __restrict__ cm3, const float* __restrict__ cv3, const float* __restrict__ en3,
                             float* __restrict__ codes)
{
    int i = blockIdx.x * blockDim.x + threadIdx.x;
    if (i >= 2320) return;
    const float *cm, *cv, *en; int j;
    if (i < 160)       { cm = cm0; cv = cv0; en = en0; j = i; }
    else if (i < 1160) { cm = cm1; cv = cv1; en = en1; j = i - 160; }
    else if (i < 2160) { cm = cm2; cv = cv2; en = en2; j = i - 1160; }
    else               { cm = cm3; cv = cv3; en = en3; j = i - 2160; }
    codes[i] = cm[j] + en[j] * expf(cv[j]);
}

// ---------------------------------------------------------------------------
// Hypernetwork: 20 -> 36 -> 36 -> 1 MLP applied to code pairs.
// Output layout: W0[16*100] | W1[100*100] | W2[100*16] | b0[100] | b1[100] | b2[16]
// ---------------------------------------------------------------------------
__global__ void hyper_kernel(const float* __restrict__ codes,
                             const float* __restrict__ w1, const float* __restrict__ bw1,
                             const float* __restrict__ w2, const float* __restrict__ bw2,
                             const float* __restrict__ w3, const float* __restrict__ bw3,
                             float* __restrict__ outw)
{
    int idx = blockIdx.x * blockDim.x + threadIdx.x;
    if (idx >= 13416) return;
    const float* c0 = codes;
    const float* c1 = codes + 160;
    const float* c2 = codes + 1160;
    const float* c3 = codes + 2160;
    float in[20];
    if (idx < 1600) {                       // W0: ci=c0 (16), cj=c1 (100)
        int a = idx / 100, b = idx % 100;
        for (int d = 0; d < 10; d++) { in[d] = c1[b*10+d]; in[10+d] = c0[a*10+d]; }
    } else if (idx < 11600) {               // W1: ci=c1, cj=c2
        int t = idx - 1600; int a = t / 100, b = t % 100;
        for (int d = 0; d < 10; d++) { in[d] = c2[b*10+d]; in[10+d] = c1[a*10+d]; }
    } else if (idx < 13200) {               // W2: ci=c2, cj=c3
        int t = idx - 11600; int a = t / 16, b = t % 16;
        for (int d = 0; d < 10; d++) { in[d] = c3[b*10+d]; in[10+d] = c2[a*10+d]; }
    } else if (idx < 13300) {               // b0: [0, c1[b]]
        int b = idx - 13200;
        for (int d = 0; d < 10; d++) { in[d] = 0.f; in[10+d] = c1[b*10+d]; }
    } else if (idx < 13400) {               // b1: [0, c2[b]]
        int b = idx - 13300;
        for (int d = 0; d < 10; d++) { in[d] = 0.f; in[10+d] = c2[b*10+d]; }
    } else {                                // b2: [0, c3[b]]
        int b = idx - 13400;
        for (int d = 0; d < 10; d++) { in[d] = 0.f; in[10+d] = c3[b*10+d]; }
    }
    float h1[36], h2[36];
    for (int j = 0; j < 36; j++) {
        float s = bw1[j];
        for (int k = 0; k < 20; k++) s += in[k] * w1[j*20+k];
        h1[j] = fmaxf(s, 0.f);
    }
    for (int j = 0; j < 36; j++) {
        float s = bw2[j];
        for (int k = 0; k < 36; k++) s += h1[k] * w2[j*36+k];
        h2[j] = fmaxf(s, 0.f);
    }
    float o = bw3[0];
    for (int k = 0; k < 36; k++) o += h2[k] * w3[k];
    outw[idx] = o;
}

// ---------------------------------------------------------------------------
// Naive fp32 direct convolution (encoder only, tiny). act: 0=none, 1=tanh
// ---------------------------------------------------------------------------
__global__ void conv_fwd(const float* __restrict__ in, const float* __restrict__ w,
                         const float* __restrict__ bias, float* __restrict__ out,
                         int Bn, int IC, int IH, int IW, int OC, int KS, int S, int P,
                         int OH, int OW, int bstr, int cstr, int act)
{
    int idx = blockIdx.x * blockDim.x + threadIdx.x;
    int total = Bn * OC * OH * OW;
    if (idx >= total) return;
    int ox = idx % OW; int t = idx / OW;
    int oy = t % OH;  t /= OH;
    int oc = t % OC;  int b = t / OC;
    float s = bias ? bias[oc] : 0.f;
    for (int ic = 0; ic < IC; ic++)
        for (int ky = 0; ky < KS; ky++) {
            int iy = oy * S + ky - P;
            if (iy < 0 || iy >= IH) continue;
            for (int kx = 0; kx < KS; kx++) {
                int ix = ox * S + kx - P;
                if (ix < 0 || ix >= IW) continue;
                s += in[b*bstr + ic*cstr + iy*IW + ix] * w[((oc*IC+ic)*KS+ky)*KS+kx];
            }
        }
    if (act == 1) s = tanhf(s);
    out[idx] = s;
}

// Per-channel sum / sumsq for fp32 NCHW, one block per channel (no atomics).
__global__ void chan_stats_f32(const float* __restrict__ y, float* __restrict__ stats,
                               int Bn, int C, int HW)
{
    int c = blockIdx.x;
    __shared__ float sb[256], sb2[256];
    float s = 0.f, ss = 0.f;
    int n = Bn * HW;
    for (int i = threadIdx.x; i < n; i += 256) {
        int b = i / HW, r = i % HW;
        float v = y[(size_t)(b*C + c)*HW + r];
        s += v; ss += v * v;
    }
    sb[threadIdx.x] = s; sb2[threadIdx.x] = ss;
    __syncthreads();
    for (int st = 128; st > 0; st >>= 1) {
        if (threadIdx.x < st) { sb[threadIdx.x] += sb[threadIdx.x+st]; sb2[threadIdx.x] += sb2[threadIdx.x+st]; }
        __syncthreads();
    }
    if (threadIdx.x == 0) { stats[2*c] = sb[0]; stats[2*c+1] = sb2[0]; }
}

__global__ void bn_relu_f32(float* __restrict__ y, const float* __restrict__ stats,
                            const float* __restrict__ g, const float* __restrict__ bb,
                            int C, int HW, int total, float invcnt)
{
    int idx = blockIdx.x * blockDim.x + threadIdx.x;
    if (idx >= total) return;
    int c = (idx / HW) % C;
    float mean = stats[2*c] * invcnt;
    float var  = stats[2*c+1] * invcnt - mean * mean;
    float v = (y[idx] - mean) * rsqrtf(var + 1e-5f) * g[c] + bb[c];
    y[idx] = fmaxf(v, 0.f);
}

// Dense fp32 fully-connected: out[b,o] = act(bias[o] + in[b,:] . w[o,:])
__global__ void fc_kernel(const float* __restrict__ in, const float* __restrict__ w,
                          const float* __restrict__ bias, float* __restrict__ out,
                          int Bn, int IN, int OUT, int relu)
{
    int idx = blockIdx.x * blockDim.x + threadIdx.x;
    if (idx >= Bn * OUT) return;
    int o = idx % OUT, b = idx / OUT;
    const float* ip = in + (size_t)b * IN;
    const float* wp = w + (size_t)o * IN;
    float s = bias[o];
    for (int k = 0; k < IN; k++) s += ip[k] * wp[k];
    out[idx] = relu ? fmaxf(s, 0.f) : s;
}

__global__ void z0_kernel(const float* __restrict__ qm, const float* __restrict__ qlv,
                          const float* __restrict__ eps, float* __restrict__ z0, int n)
{
    int i = blockIdx.x * blockDim.x + threadIdx.x;
    if (i < n) z0[i] = qm[i] + eps[i] * expf(qlv[i]);
}

// ---------------------------------------------------------------------------
// Sequential RK4 neural-ODE scan. One block per batch element (32 blocks).
// Writes zt as f16, (B*T) x 32 (K padded 16 -> 32 with zeros for fc3 WMMA).
// ---------------------------------------------------------------------------
__global__ void __launch_bounds__(128)
ode_kernel(const float* __restrict__ z0, const float* __restrict__ Wb,
           _Float16* __restrict__ zt, int T)
{
    __shared__ float W0[1600], W1[10000], W2[1600], b0[100], b1[100], b2[16];
    __shared__ float z[16], h1[100], h2[100], kv[16], ac[16], ztmp[16];
    const int b = blockIdx.x, tid = threadIdx.x;
    for (int i = tid; i < 1600;  i += 128) W0[i] = Wb[i];
    for (int i = tid; i < 10000; i += 128) W1[i] = Wb[1600 + i];
    for (int i = tid; i < 1600;  i += 128) W2[i] = Wb[11600 + i];
    for (int i = tid; i < 100;   i += 128) { b0[i] = Wb[13200 + i]; b1[i] = Wb[13300 + i]; }
    if (tid < 16) { b2[tid] = Wb[13400 + tid]; z[tid] = z0[b*16 + tid]; }
    __syncthreads();
    if (tid < 32) zt[(size_t)(b*T)*32 + tid] = (tid < 16) ? (_Float16)z[tid] : (_Float16)0.f;

#define FEVAL(SRC)                                                                          \
    if (tid < 100) { float s_ = b0[tid];                                                    \
        for (int k2 = 0; k2 < 16;  k2++) s_ += SRC[k2] * W0[k2*100 + tid];                  \
        h1[tid] = tanhf(s_); }                                                              \
    __syncthreads();                                                                        \
    if (tid < 100) { float s_ = b1[tid];                                                    \
        for (int k2 = 0; k2 < 100; k2++) s_ += h1[k2] * W1[k2*100 + tid];                   \
        h2[tid] = tanhf(s_); }                                                              \
    __syncthreads();                                                                        \
    if (tid < 16)  { float s_ = b2[tid];                                                    \
        for (int k2 = 0; k2 < 100; k2++) s_ += h2[k2] * W2[k2*16 + tid];                    \
        kv[tid] = s_; }                                                                     \
    __syncthreads();

    for (int t = 1; t < T; t++) {
        FEVAL(z)
        if (tid < 16) { ac[tid] = kv[tid];        ztmp[tid] = z[tid] + 0.05f * kv[tid]; } __syncthreads();
        FEVAL(ztmp)
        if (tid < 16) { ac[tid] += 2.f * kv[tid]; ztmp[tid] = z[tid] + 0.05f * kv[tid]; } __syncthreads();
        FEVAL(ztmp)
        if (tid < 16) { ac[tid] += 2.f * kv[tid]; ztmp[tid] = z[tid] + 0.1f  * kv[tid]; } __syncthreads();
        FEVAL(ztmp)
        if (tid < 16) { ac[tid] += kv[tid];       z[tid]   = z[tid] + (0.1f/6.f) * ac[tid]; } __syncthreads();
        if (tid < 32) zt[(size_t)(b*T + t)*32 + tid] = (tid < 16) ? (_Float16)z[tid] : (_Float16)0.f;
    }
#undef FEVAL
}

// ---------------------------------------------------------------------------
// Weight packing: convT weights (IC, OC, k, k) fp32 -> f16 B-matrix [K x Npad]
// ---------------------------------------------------------------------------
__global__ void pack_convw(const float* __restrict__ w, _Float16* __restrict__ Bm,
                           int IC, int OC, int KS, int Npad)
{
    int idx = blockIdx.x * blockDim.x + threadIdx.x;
    int total = IC * KS * KS * Npad;
    if (idx >= total) return;
    int oc = idx % Npad; int kk = idx / Npad;
    int ic = kk / (KS*KS); int r2 = kk - ic*KS*KS;
    int ky = r2 / KS, kx = r2 - ky*KS;
    Bm[idx] = (oc < OC) ? (_Float16)w[((size_t)(ic*OC + oc)*KS + ky)*KS + kx] : (_Float16)0.f;
}

// fc3_w (1024,16) -> B [32 x 1024], rows 16..31 zero (K padded).
__global__ void pack_fc3(const float* __restrict__ w, _Float16* __restrict__ Bm)
{
    int idx = blockIdx.x * blockDim.x + threadIdx.x;
    if (idx >= 32 * 1024) return;
    int k = idx / 1024, j = idx % 1024;
    Bm[idx] = (k < 16) ? (_Float16)w[j*16 + k] : (_Float16)0.f;
}

// ---------------------------------------------------------------------------
// Unified WMMA GEMM:  C[M,N] = A[M,K] * B[K,N]   (f16 in, f32 accumulate)
//   Block = 128 threads = 4 waves; block tile 128(M) x 32(N); K step 32.
//   Each wave computes a 2x2 grid of 16x16 tiles (4 v_wmma per K step,
//   each A/B fragment reused twice).
//   a_mode 0: dense A (row-major, lda)
//   a_mode 1: implicit im2col of transposed conv:
//             A[m=(n,oy,ox), kk=(ic,ky,kx)] = x[n,ic,(oy+P-ky)/S,(ox+P-kx)/S]
//             when divisible & in range else 0.  S must be a power of two.
//             The (ic,ky,kx) walk is wave-uniform -> stays in SALU.
//   c_mode 0: f16 dense store (+bias per column)
//   c_mode 1: f16 scatter to NCHW (columns >= OC dropped)
//   c_mode 2: sigmoid(f32) store of column 0 only (+bias[0])
// Fragment layouts follow CDNA5 ISA 7.12.2 (wave32).
// ---------------------------------------------------------------------------
__global__ void __launch_bounds__(128)
gemm_wmma(const _Float16* __restrict__ A, const _Float16* __restrict__ Bm,
          _Float16* __restrict__ C16, float* __restrict__ C32,
          const float* __restrict__ bias,
          int M, int N, int K,
          int a_mode, int lda,
          int IC, int IH, int IW, int KS, int S, int P, int OH, int OW, int OC,
          int c_mode)
{
    __shared__ _Float16 ldsA[128 * 32];  // [row 0..127][k 0..31]
    __shared__ _Float16 ldsB[32 * 32];   // transposed: [n 0..31][k 0..31]
    const int tid  = threadIdx.x;
    const int lane = tid & 31;
    const int wave = tid >> 5;
    const int mblk = blockIdx.x * 128;
    const int nblk = blockIdx.y * 32;
    const int HWo  = OH * OW;
    const int KK   = KS * KS;
    const int smask = S - 1;            // S in {1,2}
    const int ssh   = (S == 2) ? 1 : 0;
    const int IHW   = IH * IW;

    // ---- per-thread A-staging coordinates: one full row per thread ----
    const int m_stage = mblk + tid;
    int oyP = 0, oxP = 0;
    size_t abase = 0;                   // (nimg*IC)*IH*IW
    if (a_mode == 1) {
        int nimg = m_stage / HWo;
        int rem  = m_stage - nimg * HWo;
        int oy = rem / OW, ox = rem - oy * OW;
        oyP = oy + P; oxP = ox + P;
        abase = (size_t)nimg * IC * IHW;
    }
    // B staging: thread -> (k row, 8 consecutive n)
    const int bk  = tid >> 2;
    const int bn0 = (tid & 3) * 8;

    const int hf = lane >> 4;           // K-half served by this lane
    const int nn = lane & 15;
    const int ra0 = (wave*32 + nn) * 32;        // LDS row offsets (halves)
    const int ra1 = (wave*32 + 16 + nn) * 32;

    v8f acc00 = {}, acc01 = {}, acc10 = {}, acc11 = {};

    for (int k0 = 0; k0 < K; k0 += 32) {
        // ---- stage A tile (128 x 32 halves) ----
        if (a_mode == 0) {
            const v8h* src = (const v8h*)(A + (size_t)m_stage * lda + k0);
            v8h* dst = (v8h*)&ldsA[tid * 32];
#pragma unroll
            for (int q = 0; q < 4; q++) dst[q] = src[q];
        } else {
            // wave-uniform (ic,ky,kx) walk: decompose once, then increment
            int ic = k0 / KK;
            int r2 = k0 - ic * KK;
            int ky = r2 / KS, kx = r2 - ky * KS;
            size_t icbase = abase + (size_t)ic * IHW;
#pragma unroll 1
            for (int c = 0; c < 32; c++) {
                int ty = oyP - ky, tx = oxP - kx;
                _Float16 v = (_Float16)0.f;
                if (((ty | tx) >= 0) && ((ty & smask) == 0) && ((tx & smask) == 0)) {
                    int iy = ty >> ssh, ix = tx >> ssh;
                    if (iy < IH && ix < IW) v = A[icbase + iy * IW + ix];
                }
                ldsA[tid * 32 + c] = v;
                if (++kx == KS) { kx = 0;
                    if (++ky == KS) { ky = 0; ic++; icbase += IHW; } }
            }
        }
        // ---- stage B tile transposed (32 x 32 halves) ----
        {
            v8h gl = *(const v8h*)&Bm[(size_t)(k0 + bk) * N + nblk + bn0];
#pragma unroll
            for (int j = 0; j < 8; j++) ldsB[(bn0 + j) * 32 + bk] = gl[j];
        }
        if (k0 + 32 < K)
            __builtin_prefetch(&Bm[(size_t)(k0 + 32) * N + nblk], 0, 1);
        __syncthreads();

        // ---- build fragments per CDNA5 wave32 layouts ----
        v8h a00 = *(const v8h*)&ldsA[ra0 + hf*8];       // K = hf*8 + 0..7
        v8h a01 = *(const v8h*)&ldsA[ra0 + 16 + hf*8];  // K = 16 + hf*8 + 0..7
        v8h a10 = *(const v8h*)&ldsA[ra1 + hf*8];
        v8h a11 = *(const v8h*)&ldsA[ra1 + 16 + hf*8];
        v8h bl0 = *(const v8h*)&ldsB[nn*32 + hf*8];        // cols 0..15
        v8h bh0 = *(const v8h*)&ldsB[nn*32 + 16 + hf*8];
        v8h bl1 = *(const v8h*)&ldsB[(16 + nn)*32 + hf*8]; // cols 16..31
        v8h bh1 = *(const v8h*)&ldsB[(16 + nn)*32 + 16 + hf*8];
        v16h af0, af1, bf0, bf1;
#pragma unroll
        for (int j = 0; j < 8; j++) {
            af0[j] = a00[j]; af0[j+8] = a01[j];
            af1[j] = a10[j]; af1[j+8] = a11[j];
            bf0[j] = bl0[j]; bf0[j+8] = bh0[j];
            bf1[j] = bl1[j]; bf1[j+8] = bh1[j];
        }
        acc00 = __builtin_amdgcn_wmma_f32_16x16x32_f16(false, af0, false, bf0, (short)0, acc00, false, false);
        acc01 = __builtin_amdgcn_wmma_f32_16x16x32_f16(false, af0, false, bf1, (short)0, acc01, false, false);
        acc10 = __builtin_amdgcn_wmma_f32_16x16x32_f16(false, af1, false, bf0, (short)0, acc10, false, false);
        acc11 = __builtin_amdgcn_wmma_f32_16x16x32_f16(false, af1, false, bf1, (short)0, acc11, false, false);
        __syncthreads();
    }

    // ---- epilogue: tile (tm,tn); C layout: VGPR i -> M = i + 8*hf, N = nn ----
    v8f* accs[2][2] = { { &acc00, &acc01 }, { &acc10, &acc11 } };
#pragma unroll
    for (int tm = 0; tm < 2; tm++) {
#pragma unroll
        for (int tn = 0; tn < 2; tn++) {
            v8f& a = *accs[tm][tn];
            const int n = nblk + tn*16 + nn;
            const int mbase = mblk + wave*32 + tm*16 + hf*8;
            if (c_mode == 0) {
                float bv = bias ? bias[n] : 0.f;
#pragma unroll
                for (int i = 0; i < 8; i++)
                    C16[(size_t)(mbase + i) * N + n] = (_Float16)(a[i] + bv);
            } else if (c_mode == 1) {
                if (n < OC) {
                    int ni  = mbase / HWo;
                    int rem = mbase - ni * HWo;
#pragma unroll
                    for (int i = 0; i < 8; i++) {
                        C16[((size_t)ni * OC + n) * HWo + rem] = (_Float16)a[i];
                        if (++rem == HWo) { rem = 0; ni++; }
                    }
                }
            } else {
                if (n == 0) {
                    float bv = bias[0];
#pragma unroll
                    for (int i = 0; i < 8; i++) {
                        float v = a[i] + bv;
                        C32[mbase + i] = 1.f / (1.f + expf(-v));
                    }
                }
            }
        }
    }
}

// Per-channel stats over f16 NCHW with grid (C, slabs); atomic accumulate.
__global__ void chan_stats_f16(const _Float16* __restrict__ y, float* __restrict__ stats,
                               int Nimg, int C, int HW)
{
    int c = blockIdx.x;
    __shared__ float sb[256], sb2[256];
    float s = 0.f, ss = 0.f;
    for (int n = blockIdx.y; n < Nimg; n += gridDim.y) {
        const _Float16* p = y + ((size_t)n * C + c) * HW;
        for (int i = threadIdx.x; i < HW; i += 256) {
            float v = (float)p[i]; s += v; ss += v * v;
        }
    }
    sb[threadIdx.x] = s; sb2[threadIdx.x] = ss;
    __syncthreads();
    for (int st = 128; st > 0; st >>= 1) {
        if (threadIdx.x < st) { sb[threadIdx.x] += sb[threadIdx.x+st]; sb2[threadIdx.x] += sb2[threadIdx.x+st]; }
        __syncthreads();
    }
    if (threadIdx.x == 0) { atomicAdd(&stats[2*c], sb[0]); atomicAdd(&stats[2*c+1], sb2[0]); }
}

__global__ void bn_relu_f16(_Float16* __restrict__ y, const float* __restrict__ stats,
                            const float* __restrict__ g, const float* __restrict__ bb,
                            int C, int HW, long long total, float invcnt)
{
    long long idx = (long long)blockIdx.x * blockDim.x + threadIdx.x;
    if (idx >= total) return;
    int c = (int)((idx / HW) % C);
    float mean = stats[2*c] * invcnt;
    float var  = stats[2*c+1] * invcnt - mean * mean;
    float v = ((float)y[idx] - mean) * rsqrtf(var + 1e-5f) * g[c] + bb[c];
    y[idx] = (_Float16)fmaxf(v, 0.f);
}

// ---------------------------------------------------------------------------
// Host orchestration
// ---------------------------------------------------------------------------
static inline size_t al256(size_t v) { return (v + 255) & ~(size_t)255; }

extern "C" void kernel_launch(void* const* d_in, const int* in_sizes, int n_in,
                              void* d_out, int out_size, void* d_ws, size_t ws_size,
                              hipStream_t stream)
{
    const float* x     = (const float*)d_in[0];
    const float* cm0   = (const float*)d_in[1],  *cv0 = (const float*)d_in[2],  *en0 = (const float*)d_in[3];
    const float* cm1   = (const float*)d_in[4],  *cv1 = (const float*)d_in[5],  *en1 = (const float*)d_in[6];
    const float* cm2   = (const float*)d_in[7],  *cv2 = (const float*)d_in[8],  *en2 = (const float*)d_in[9];
    const float* cm3   = (const float*)d_in[10], *cv3 = (const float*)d_in[11], *en3 = (const float*)d_in[12];
    const float* eps_z = (const float*)d_in[13];
    const float* hp_w1 = (const float*)d_in[14], *hp_b1 = (const float*)d_in[15];
    const float* hp_w2 = (const float*)d_in[16], *hp_b2 = (const float*)d_in[17];
    const float* hp_w3 = (const float*)d_in[18], *hp_b3 = (const float*)d_in[19];
    const float* ec_w1 = (const float*)d_in[20];
    const float* bn1_g = (const float*)d_in[22], *bn1_b = (const float*)d_in[23];
    const float* ec_w2 = (const float*)d_in[24];
    const float* bn2_g = (const float*)d_in[26], *bn2_b = (const float*)d_in[27];
    const float* ec_w3 = (const float*)d_in[28], *ec_b3 = (const float*)d_in[29];
    const float* mz_w1 = (const float*)d_in[30], *mz_b1 = (const float*)d_in[31];
    const float* mz_w2 = (const float*)d_in[32], *mz_b2 = (const float*)d_in[33];
    const float* lz_w1 = (const float*)d_in[34], *lz_b1 = (const float*)d_in[35];
    const float* lz_w2 = (const float*)d_in[36], *lz_b2 = (const float*)d_in[37];
    const float* fc3_w = (const float*)d_in[38], *fc3_b = (const float*)d_in[39];
    const float* dc_w1 = (const float*)d_in[40];
    const float* dbn1_g = (const float*)d_in[42], *dbn1_b = (const float*)d_in[43];
    const float* dc_w2 = (const float*)d_in[44];
    const float* dbn2_g = (const float*)d_in[46], *dbn2_b = (const float*)d_in[47];
    const float* dc_w3 = (const float*)d_in[48];
    const float* dbn3_g = (const float*)d_in[50], *dbn3_b = (const float*)d_in[51];
    const float* dc_w4 = (const float*)d_in[52], *dc_b4 = (const float*)d_in[53];
    (void)in_sizes; (void)n_in; (void)out_size; (void)ws_size;

    char* ws = (char*)d_ws;
    size_t off = 0;
    auto alloc = [&](size_t bytes) { size_t o = off; off = al256(off + bytes); return o; };
    size_t o_codes = alloc(2320 * 4);
    size_t o_hyper = alloc(13416 * 4);
    size_t o_e1    = alloc((size_t)32*16*16*16*4);
    size_t o_e2    = alloc((size_t)32*32*8*8*4);
    size_t o_henc  = alloc((size_t)32*1024*4);
    size_t o_hid   = alloc((size_t)2*32*2048*4);
    size_t o_qm    = alloc(32*16*4);
    size_t o_qlv   = alloc(32*16*4);
    size_t o_z0    = alloc(32*16*4);
    size_t o_stats = alloc(544 * 4);          // enc1(32) enc2(64) d1(256) d2(128) d3(64) floats
    size_t o_zt    = alloc((size_t)2048*32*2);
    size_t o_fc3B  = alloc((size_t)32*1024*2);
    size_t o_s     = alloc((size_t)2048*1024*2);
    size_t o_w1B   = alloc((size_t)1024*128*2);
    size_t o_w2B   = alloc((size_t)3200*64*2);
    size_t o_w3B   = alloc((size_t)1600*32*2);
    size_t o_w4B   = alloc((size_t)800*32*2);
    size_t o_y1    = alloc((size_t)2048*128*49*2);
    size_t o_y2    = alloc((size_t)2048*64*225*2);
    size_t o_y3    = alloc((size_t)2048*32*1024*2);

    float* codes = (float*)(ws + o_codes);
    float* hyper = (float*)(ws + o_hyper);
    float* e1    = (float*)(ws + o_e1);
    float* e2    = (float*)(ws + o_e2);
    float* henc  = (float*)(ws + o_henc);
    float* hidmz = (float*)(ws + o_hid);
    float* hidlz = hidmz + (size_t)32*2048;
    float* qm    = (float*)(ws + o_qm);
    float* qlv   = (float*)(ws + o_qlv);
    float* z0b   = (float*)(ws + o_z0);
    float* stats = (float*)(ws + o_stats);
    float* st_e1 = stats;        // 16 ch
    float* st_e2 = stats + 32;   // 32 ch
    float* st_d1 = stats + 96;   // 128 ch
    float* st_d2 = stats + 352;  // 64 ch
    float* st_d3 = stats + 480;  // 32 ch
    _Float16* ztb  = (_Float16*)(ws + o_zt);
    _Float16* fc3B = (_Float16*)(ws + o_fc3B);
    _Float16* sbuf = (_Float16*)(ws + o_s);
    _Float16* w1B  = (_Float16*)(ws + o_w1B);
    _Float16* w2B  = (_Float16*)(ws + o_w2B);
    _Float16* w3B  = (_Float16*)(ws + o_w3B);
    _Float16* w4B  = (_Float16*)(ws + o_w4B);
    _Float16* y1   = (_Float16*)(ws + o_y1);
    _Float16* y2   = (_Float16*)(ws + o_y2);
    _Float16* y3   = (_Float16*)(ws + o_y3);
    float* outp    = (float*)d_out;

    auto NB = [](long long n, int b) { return (unsigned)((n + b - 1) / b); };

    hipMemsetAsync(stats, 0, 544 * 4, stream);

    // ---- hyper path ----
    codes_kernel<<<NB(2320,256),256,0,stream>>>(cm0,cv0,en0, cm1,cv1,en1, cm2,cv2,en2, cm3,cv3,en3, codes);
    hyper_kernel<<<NB(13416,256),256,0,stream>>>(codes, hp_w1,hp_b1, hp_w2,hp_b2, hp_w3,hp_b3, hyper);

    // ---- encoder (fp32; conv biases before BN cancel) ----
    conv_fwd<<<NB(131072,256),256,0,stream>>>(x, ec_w1, nullptr, e1, 32,5,32,32, 16,5,2,2, 16,16, 65536,1024, 0);
    chan_stats_f32<<<16,256,0,stream>>>(e1, st_e1, 32,16,256);
    bn_relu_f32<<<NB(131072,256),256,0,stream>>>(e1, st_e1, bn1_g, bn1_b, 16,256, 131072, 1.f/8192.f);
    conv_fwd<<<NB(65536,256),256,0,stream>>>(e1, ec_w2, nullptr, e2, 32,16,16,16, 32,5,2,2, 8,8, 4096,256, 0);
    chan_stats_f32<<<32,256,0,stream>>>(e2, st_e2, 32,32,64);
    bn_relu_f32<<<NB(65536,256),256,0,stream>>>(e2, st_e2, bn2_g, bn2_b, 32,64, 65536, 1.f/2048.f);
    conv_fwd<<<NB(32768,256),256,0,stream>>>(e2, ec_w3, ec_b3, henc, 32,32,8,8, 64,5,2,2, 4,4, 2048,64, 1);

    // ---- variational heads + z0 ----
    fc_kernel<<<NB(65536,256),256,0,stream>>>(henc, mz_w1, mz_b1, hidmz, 32,1024,2048, 1);
    fc_kernel<<<NB(65536,256),256,0,stream>>>(henc, lz_w1, lz_b1, hidlz, 32,1024,2048, 1);
    fc_kernel<<<NB(512,256),256,0,stream>>>(hidmz, mz_w2, mz_b2, qm,  32,2048,16, 0);
    fc_kernel<<<NB(512,256),256,0,stream>>>(hidlz, lz_w2, lz_b2, qlv, 32,2048,16, 0);
    z0_kernel<<<NB(512,256),256,0,stream>>>(qm, qlv, eps_z, z0b, 512);

    // ---- sequential RK4 scan -> zt (2048 x 32 f16, K padded) ----
    ode_kernel<<<32,128,0,stream>>>(z0b, hyper, ztb, 64);

    // ---- fc3: s = zt @ fc3_w.T + b   (WMMA, dense A, M=2048 N=1024 K=32) ----
    pack_fc3<<<NB(32768,256),256,0,stream>>>(fc3_w, fc3B);
    gemm_wmma<<<dim3(2048/128, 1024/32),128,0,stream>>>(ztb, fc3B, sbuf, nullptr, fc3_b,
        2048,1024,32, 0,32, 0,0,0,1,1,0,0,0,0, 0);

    // ---- decoder convT1: (2048,64,4,4)->(2048,128,7,7), K=1024 ----
    pack_convw<<<NB(131072,256),256,0,stream>>>(dc_w1, w1B, 64,128,4,128);
    gemm_wmma<<<dim3(100352/128, 128/32),128,0,stream>>>(sbuf, w1B, y1, nullptr, nullptr,
        100352,128,1024, 1,0, 64,4,4,4,1,0, 7,7,128, 1);
    chan_stats_f16<<<dim3(128,32),256,0,stream>>>(y1, st_d1, 2048,128,49);
    bn_relu_f16<<<NB(12845056,256),256,0,stream>>>(y1, st_d1, dbn1_g, dbn1_b, 128,49, 12845056LL, 1.f/100352.f);

    // ---- decoder convT2: ->(2048,64,15,15), K=3200 ----
    pack_convw<<<NB(204800,256),256,0,stream>>>(dc_w2, w2B, 128,64,5,64);
    gemm_wmma<<<dim3(460800/128, 64/32),128,0,stream>>>(y1, w2B, y2, nullptr, nullptr,
        460800,64,3200, 1,0, 128,7,7,5,2,1, 15,15,64, 1);
    chan_stats_f16<<<dim3(64,32),256,0,stream>>>(y2, st_d2, 2048,64,225);
    bn_relu_f16<<<NB(29491200,256),256,0,stream>>>(y2, st_d2, dbn2_g, dbn2_b, 64,225, 29491200LL, 1.f/460800.f);

    // ---- decoder convT3: ->(2048,32,32,32), K=1600 ----
    pack_convw<<<NB(51200,256),256,0,stream>>>(dc_w3, w3B, 64,32,5,32);
    gemm_wmma<<<dim3(2097152/128, 1),128,0,stream>>>(y2, w3B, y3, nullptr, nullptr,
        2097152,32,1600, 1,0, 64,15,15,5,2,1, 32,32,32, 1);
    chan_stats_f16<<<dim3(32,32),256,0,stream>>>(y3, st_d3, 2048,32,1024);
    bn_relu_f16<<<NB(67108864,256),256,0,stream>>>(y3, st_d3, dbn3_g, dbn3_b, 32,1024, 67108864LL, 1.f/2097152.f);

    // ---- decoder convT4 + sigmoid -> d_out (f32), K=800, N padded 1->32 ----
    pack_convw<<<NB(25600,256),256,0,stream>>>(dc_w4, w4B, 32,1,5,32);
    gemm_wmma<<<dim3(2097152/128, 1),128,0,stream>>>(y3, w4B, nullptr, outp, dc_b4,
        2097152,32,800, 1,0, 32,32,32,5,1,2, 32,32,1, 2);
}
